// EMLConsensusField_3332894621817
// MI455X (gfx1250) — matrix operand order repaired and checked by reading, not compiled, not run on hardware
//
#include <hip/hip_runtime.h>
#include <hip/hip_bf16.h>

#define SIDE  48
#define PP    (SIDE*SIDE)        // 2304
#define BB    2
#define K9    9
#define FDIM  32
#define HIDN  64
#define NHYP  4
#define RDIM  8
#define JNT   104
#define JPAD  128
#define GROUPS (BB*PP)           // 4608
#define WPB   4                  // waves per block
#define NOUT  (BB*PP*NHYP)       // 18432

typedef __attribute__((ext_vector_type(16))) _Float16 v16h;
typedef __attribute__((ext_vector_type(8)))  float    v8f;

__device__ __forceinline__ float gelu_exact(float x){
    return 0.5f * x * (1.0f + erff(x * 0.7071067811865475f));
}
__device__ __forceinline__ float softplus_f(float x){
    return (x > 20.f) ? x : log1pf(expf(x));
}
__device__ __forceinline__ unsigned int pack2h(float lo, float hi){
    union { _Float16 h[2]; unsigned int u; } t;
    t.h[0] = (_Float16)lo; t.h[1] = (_Float16)hi;
    return t.u;
}

__global__ void eml_zero_scalars(float* out){
    if (threadIdx.x < 3) out[7 * NOUT + threadIdx.x] = 0.f;
}

__global__ __launch_bounds__(128)
void eml_main(const float* __restrict__ hyp,
              const float* __restrict__ activation,
              const float* __restrict__ drive,
              const float* __restrict__ resist,
              const float* __restrict__ relative,
              const float* __restrict__ ln_w,  const float* __restrict__ ln_b,
              const float* __restrict__ sup_w1, const float* __restrict__ sup_b1,
              const float* __restrict__ sup_w2, const float* __restrict__ sup_b2,
              const float* __restrict__ con_w1, const float* __restrict__ con_b1,
              const float* __restrict__ con_w2, const float* __restrict__ con_b2,
              const float* __restrict__ eml_bias_p,
              float* __restrict__ out)
{
    // W1 weights pre-swizzled into exact B-fragment layout:
    // Wf[mlp][ks][t][lane][r] packs W[k0][n],W[k0+1][n] with
    //   n = t*16 + (lane&15), k0 = ks*32 + ((lane>=16)?16:0) + 2r
    __shared__ __align__(16) unsigned int Wf[2][4][4][32][8];   // 32 KB
    __shared__ __align__(16) _Float16 Jl[WPB][16][JPAD];        // 16 KB
    __shared__ float b1s[2][HIDN], w2s[2][HIDN];
    __shared__ float lnw_s[JNT], lnb_s[JNT];
    __shared__ float rel_s[K9 * RDIM];
    __shared__ float vals[WPB][2][16];
    __shared__ float accv[WPB][2][NHYP];
    __shared__ float gmass[WPB];

    const int tid = threadIdx.x;

    for (int i = tid; i < 2*4*4*32*8; i += 128) {
        const int r   =  i        & 7;
        const int ln  = (i >> 3)  & 31;
        const int t   = (i >> 8)  & 3;
        const int ks  = (i >> 10) & 3;
        const int mlp = (i >> 12) & 1;
        const int n   = t*16 + (ln & 15);
        const int k0  = ks*32 + ((ln >= 16) ? 16 : 0) + 2*r;
        const float* W = mlp ? con_w1 : sup_w1;
        const float lo = (k0     < JNT) ? W[(size_t)k0      * HIDN + n] : 0.f;
        const float hi = (k0 + 1 < JNT) ? W[(size_t)(k0+1)  * HIDN + n] : 0.f;
        Wf[mlp][ks][t][ln][r] = pack2h(lo, hi);
    }
    for (int i = tid; i < HIDN; i += 128) {
        b1s[0][i] = sup_b1[i]; w2s[0][i] = sup_w2[i];
        b1s[1][i] = con_b1[i]; w2s[1][i] = con_w2[i];
    }
    for (int i = tid; i < JNT; i += 128) { lnw_s[i] = ln_w[i]; lnb_s[i] = ln_b[i]; }
    for (int i = tid; i < K9*RDIM; i += 128) rel_s[i] = relative[i];
    __syncthreads();

    const int wave = tid >> 5;
    const int lane = tid & 31;
    const int m    = lane & 15;           // pair within tile / WMMA row or column
    const int half = lane >> 4;
    const int group = blockIdx.x * WPB + wave;
    const int b  = group / PP;
    const int p  = group % PP;
    const int py = p / SIDE, px = p % SIDE;

    if (lane < NHYP) { accv[wave][0][lane] = 0.f; accv[wave][1][lane] = 0.f; }
    if (lane == 31)  gmass[wave] = 0.f;
    const float b2v0   = sup_b2[0];
    const float b2v1   = con_b2[0];
    const float embias = eml_bias_p[0];

    for (int tix = 0; tix < 9; ++tix) {
        // ---- decode this lane's pair (l = 0..143 within the (b,p) group)
        const int l   = tix*16 + m;
        const int h   = l / 36;
        const int rem = l % 36;
        const int k   = rem >> 2;
        const int h2  = rem & 3;
        const int ny  = py + k/3 - 1, nx = px + k%3 - 1;
        const bool valid = (ny >= 0) & (ny < SIDE) & (nx >= 0) & (nx < SIDE);
        const int  np = valid ? (ny*SIDE + nx) : p;
        const float vm = valid ? 1.f : 0.f;

        const float* tgt = hyp + (size_t)((b*PP + p )*NHYP + h ) * FDIM;
        const float* src = hyp + (size_t)((b*PP + np)*NHYP + h2) * FDIM;
        const float* rl  = rel_s + k * RDIM;

        float tg[FDIM], sm[FDIM];
        {
            const float4* t4 = reinterpret_cast<const float4*>(tgt);
            const float4* s4 = reinterpret_cast<const float4*>(src);
            #pragma unroll
            for (int i = 0; i < 8; ++i) {
                float4 a = t4[i], c = s4[i];
                tg[4*i+0]=a.x; tg[4*i+1]=a.y; tg[4*i+2]=a.z; tg[4*i+3]=a.w;
                sm[4*i+0]=vm*c.x; sm[4*i+1]=vm*c.y; sm[4*i+2]=vm*c.z; sm[4*i+3]=vm*c.w;
            }
        }
        // ---- layernorm stats over all 104 features (computed redundantly per lane)
        float s = 0.f, sq = 0.f;
        #pragma unroll
        for (int d = 0; d < FDIM; ++d) { float v = tg[d];        s += v; sq += v*v; }
        #pragma unroll
        for (int d = 0; d < FDIM; ++d) { float v = sm[d];        s += v; sq += v*v; }
        #pragma unroll
        for (int d = 0; d < FDIM; ++d) { float v = tg[d]-sm[d];  s += v; sq += v*v; }
        #pragma unroll
        for (int r = 0; r < RDIM; ++r) { float v = rl[r];        s += v; sq += v*v; }
        const float mu   = s * (1.f / JNT);
        const float rstd = rsqrtf(fmaxf(sq * (1.f / JNT) - mu*mu, 0.f) + 1e-5f);

        // ---- normalized f16 joint row in LDS (each lane writes its 52-feature half)
        _Float16* Jh = &Jl[wave][m][0];
        if (half == 0) {
            #pragma unroll
            for (int d = 0; d < 32; ++d)
                Jh[d]      = (_Float16)((tg[d]      - mu)*rstd*lnw_s[d]      + lnb_s[d]);
            #pragma unroll
            for (int d = 0; d < 20; ++d)
                Jh[32+d]   = (_Float16)((sm[d]      - mu)*rstd*lnw_s[32+d]   + lnb_s[32+d]);
        } else {
            #pragma unroll
            for (int d = 20; d < 32; ++d)
                Jh[32+d]   = (_Float16)((sm[d]      - mu)*rstd*lnw_s[32+d]   + lnb_s[32+d]);
            #pragma unroll
            for (int d = 0; d < 32; ++d)
                Jh[64+d]   = (_Float16)((tg[d]-sm[d]- mu)*rstd*lnw_s[64+d]   + lnb_s[64+d]);
            #pragma unroll
            for (int r = 0; r < RDIM; ++r)
                Jh[96+r]   = (_Float16)((rl[r]      - mu)*rstd*lnw_s[96+r]   + lnb_s[96+r]);
            #pragma unroll
            for (int j = JNT; j < JPAD; ++j) Jh[j] = (_Float16)0.f;
        }
        __syncthreads();

        // ---- layer 1: M16 x K128 x N64 for BOTH MLPs, A fragment shared
        v8f acc[2][4];
        #pragma unroll
        for (int q = 0; q < 2; ++q)
            #pragma unroll
            for (int t = 0; t < 4; ++t)
                #pragma unroll
                for (int i = 0; i < 8; ++i) acc[q][t][i] = 0.f;

        const unsigned int* rowu = reinterpret_cast<const unsigned int*>(&Jl[wave][m][0]);
        const int h4 = half ? 4 : 0;
        #pragma unroll
        for (int ks = 0; ks < 4; ++ks) {
            union { v16h v; unsigned int u[8]; } A;
            #pragma unroll
            for (int r = 0; r < 8; ++r)
                A.u[r] = rowu[ks*16 + 2*h4 + r];
            #pragma unroll
            for (int q = 0; q < 2; ++q) {
                #pragma unroll
                for (int t = 0; t < 4; ++t) {
                    union { v16h v; unsigned int u[8]; } Bf;
                    #pragma unroll
                    for (int r = 0; r < 8; ++r) Bf.u[r] = Wf[q][ks][t][lane][r];
                    acc[q][t] = __builtin_amdgcn_wmma_f32_16x16x32_f16(
                        false, A.v, false, Bf.v, (short)0, acc[q][t], false, false);
                }
            }
        }

        // ---- layer 2: bias + exact GELU + dot(w2), reduce over N via shfl butterfly
        #pragma unroll
        for (int q = 0; q < 2; ++q) {
            float part[8];
            #pragma unroll
            for (int r = 0; r < 8; ++r) part[r] = 0.f;
            #pragma unroll
            for (int t = 0; t < 4; ++t) {
                const int n = t*16 + m;
                const float b1v = b1s[q][n];
                const float w2v = w2s[q][n];
                #pragma unroll
                for (int r = 0; r < 8; ++r)
                    part[r] += gelu_exact(acc[q][t][r] + b1v) * w2v;
            }
            #pragma unroll
            for (int msk = 1; msk <= 8; msk <<= 1)
                #pragma unroll
                for (int r = 0; r < 8; ++r)
                    part[r] += __shfl_xor(part[r], msk, 32);
            if (m == 0) {
                const float b2 = q ? b2v1 : b2v0;
                #pragma unroll
                for (int r = 0; r < 8; ++r)
                    vals[wave][q][half*8 + r] = softplus_f(part[r] + b2);
            }
        }
        __syncthreads();

        // ---- gated accumulation (one lane per pair)
        if (half == 0) {
            const float gate = vm * activation[(size_t)(b*PP + np)*NHYP + h2];
            atomicAdd(&accv[wave][0][h], vals[wave][0][m] * gate);
            atomicAdd(&accv[wave][1][h], vals[wave][1][m] * gate);
            if (h == 0) atomicAdd(&gmass[wave], gate);
        }
        __syncthreads();
    }

    // ---- finalize per (b,p,h)
    if (lane < NHYP) {
        const int h = lane;
        const float gm  = fmaxf(gmass[wave], 1e-6f);
        const float sup = accv[wave][0][h] / gm;
        const float con = accv[wave][1][h] / gm;
        const int idx = group * NHYP + h;
        const float pd = drive[idx]  + sup;
        const float pr = resist[idx] + con;
        float en = pd - pr + embias;
        en = fminf(fmaxf(en, -3.f), 3.f);
        const float act = 1.f / (1.f + expf(-en));
        out[0*NOUT + idx] = sup;
        out[1*NOUT + idx] = con;
        out[2*NOUT + idx] = pd;
        out[3*NOUT + idx] = pr;
        out[4*NOUT + idx] = en;
        out[5*NOUT + idx] = act;
        out[6*NOUT + idx] = gm;
        const float invN = 1.f / (float)NOUT;
        const float ent  = -(act * logf(act + 1e-8f) +
                             (1.f - act) * logf(1.f - act + 1e-8f));
        atomicAdd(&out[7*NOUT + 0], act * invN);
        atomicAdd(&out[7*NOUT + 1], ent * invN);
        atomicAdd(&out[7*NOUT + 2], (act > 0.5f) ? invN : 0.f);
    }
}

extern "C" void kernel_launch(void* const* d_in, const int* in_sizes, int n_in,
                              void* d_out, int out_size, void* d_ws, size_t ws_size,
                              hipStream_t stream) {
    (void)in_sizes; (void)n_in; (void)out_size; (void)d_ws; (void)ws_size;
    const float* hyp      = (const float*)d_in[0];
    const float* act      = (const float*)d_in[1];
    const float* drive    = (const float*)d_in[2];
    const float* resist   = (const float*)d_in[3];
    const float* relative = (const float*)d_in[4];
    const float* ln_w     = (const float*)d_in[5];
    const float* ln_b     = (const float*)d_in[6];
    const float* sup_w1   = (const float*)d_in[7];
    const float* sup_b1   = (const float*)d_in[8];
    const float* sup_w2   = (const float*)d_in[9];
    const float* sup_b2   = (const float*)d_in[10];
    const float* con_w1   = (const float*)d_in[11];
    const float* con_b1   = (const float*)d_in[12];
    const float* con_w2   = (const float*)d_in[13];
    const float* con_b2   = (const float*)d_in[14];
    const float* eml_bias = (const float*)d_in[15];
    float* out = (float*)d_out;

    eml_zero_scalars<<<1, 64, 0, stream>>>(out);
    eml_main<<<GROUPS / WPB, 128, 0, stream>>>(
        hyp, act, drive, resist, relative, ln_w, ln_b,
        sup_w1, sup_b1, sup_w2, sup_b2,
        con_w1, con_b1, con_w2, con_b2, eml_bias, out);
}
